// GATom_11458972746251
// MI455X (gfx1250) — matrix operand order repaired and checked by reading, not compiled, not run on hardware
//
#include <hip/hip_runtime.h>
#include <math.h>

#define NN 100000
#define NE 1600000
#define NG 1000
#define HID 64
#define GRP 10

typedef __attribute__((ext_vector_type(16))) _Float16 v16h;
typedef __attribute__((ext_vector_type(2)))  __fp16   pkh2;   // cvt_pkrtz result type
typedef __attribute__((ext_vector_type(8)))  float    v8f;

// ---------------------------------------------------------------------------
// order-preserving float<->uint key for atomicMax-based segment max
// ---------------------------------------------------------------------------
__device__ __forceinline__ unsigned fkey(float f) {
    unsigned u = __float_as_uint(f);
    return (u & 0x80000000u) ? ~u : (u | 0x80000000u);
}
__device__ __forceinline__ float fdec(unsigned k) {
    unsigned u = (k & 0x80000000u) ? (k ^ 0x80000000u) : ~k;
    return __uint_as_float(u);
}

// ---------------------------------------------------------------------------
// WMMA GEMM: C[M, NT*16] = act(A[M,K] @ B + bias), K compile-time.
//   wTrans=0: B[k,n] = W[k*Nout + n]   (x @ W convention)
//   wTrans=1: B[k,n] = W[n*K + k]      (x @ W.T convention, GRU weights)
// f16 inputs, f32 accumulate via v_wmma_f32_16x16x32_f16.
// One wave per 16-row tile; weights staged in LDS pre-swizzled to B-fragment
// layout so each k-step is one 32B LDS vector load per N-tile. A-fragments:
// unguarded float4/float2 vector loads for full k-tiles, clamped+select for
// the single padded tail tile (no exec-mask branches, no per-load waits).
// ---------------------------------------------------------------------------
template<int NT, int K>
__global__ __launch_bounds__(128) void gemm_wmma(
    const float* __restrict__ A, const float* __restrict__ W,
    const float* __restrict__ bias, float* __restrict__ C,
    int M, int act, int wTrans)
{
    constexpr int Nout = NT * 16;
    constexpr int Kpad = (K + 31) & ~31;
    constexpr int KT = Kpad / 32;
    constexpr bool ROW16 = ((K % 4) == 0);   // row stride 16B-aligned?
    __shared__ __align__(32) _Float16 Bs[Kpad * Nout];

    // stage weights into fragment-layout LDS (zero padding to Kpad)
    for (int idx = threadIdx.x; idx < Kpad * Nout; idx += 128) {
        int k = idx / Nout, n = idx % Nout;
        float v = 0.0f;
        if (k < K) v = wTrans ? W[n * K + k] : W[k * Nout + n];
        int kt = k >> 5, kk = k & 31, nt = n >> 4, nn = n & 15;
        int lane = nn + (((kk >> 3) & 1) << 4);
        int j = (kk & 7) + ((kk >> 4) << 3);
        Bs[(((kt * NT + nt) * 32 + lane) << 4) + j] = (_Float16)v;
    }
    __syncthreads();

    int wave = threadIdx.x >> 5;
    int lane = threadIdx.x & 31;
    int row0 = (blockIdx.x * 4 + wave) * 16;
    if (row0 >= M) return;

    int mA = row0 + (lane & 15);
    if (mA > M - 1) mA = M - 1;                 // clamp; store is guarded
    const float* arow = A + (size_t)mA * K;
    int kOff = (lane >> 4) * 8;

    v8f acc[NT];
#pragma unroll
    for (int nt = 0; nt < NT; ++nt) acc[nt] = (v8f){0,0,0,0,0,0,0,0};

    union AFrag { v16h v; pkh2 h[8]; };

#pragma unroll
    for (int kt = 0; kt < KT; ++kt) {
        const int kb = kt * 32;
        AFrag af;
        if (kb + 32 <= K) {
            // fully in-bounds tile: pure vector loads, packed converts
            if (ROW16) {
                float4 f0 = *(const float4*)(arow + kb + kOff);
                float4 f1 = *(const float4*)(arow + kb + kOff + 4);
                float4 f2 = *(const float4*)(arow + kb + 16 + kOff);
                float4 f3 = *(const float4*)(arow + kb + 16 + kOff + 4);
                af.h[0] = __builtin_amdgcn_cvt_pkrtz(f0.x, f0.y);
                af.h[1] = __builtin_amdgcn_cvt_pkrtz(f0.z, f0.w);
                af.h[2] = __builtin_amdgcn_cvt_pkrtz(f1.x, f1.y);
                af.h[3] = __builtin_amdgcn_cvt_pkrtz(f1.z, f1.w);
                af.h[4] = __builtin_amdgcn_cvt_pkrtz(f2.x, f2.y);
                af.h[5] = __builtin_amdgcn_cvt_pkrtz(f2.z, f2.w);
                af.h[6] = __builtin_amdgcn_cvt_pkrtz(f3.x, f3.y);
                af.h[7] = __builtin_amdgcn_cvt_pkrtz(f3.z, f3.w);
            } else {
                // 8B-aligned rows (K=50): float2 loads
#pragma unroll
                for (int q = 0; q < 4; ++q) {
                    float2 g0 = *(const float2*)(arow + kb + kOff + 2 * q);
                    float2 g1 = *(const float2*)(arow + kb + 16 + kOff + 2 * q);
                    af.h[q]     = __builtin_amdgcn_cvt_pkrtz(g0.x, g0.y);
                    af.h[4 + q] = __builtin_amdgcn_cvt_pkrtz(g1.x, g1.y);
                }
            }
        } else {
            // tail tile: clamped addresses + select, branch-free
#pragma unroll
            for (int q = 0; q < 4; ++q) {
                int k0 = kb + kOff + 2 * q;
                int k1 = kb + 16 + kOff + 2 * q;
                float a0 = arow[k0     < K ? k0     : K - 1];
                float a1 = arow[k0 + 1 < K ? k0 + 1 : K - 1];
                float b0 = arow[k1     < K ? k1     : K - 1];
                float b1 = arow[k1 + 1 < K ? k1 + 1 : K - 1];
                a0 = k0     < K ? a0 : 0.0f;
                a1 = k0 + 1 < K ? a1 : 0.0f;
                b0 = k1     < K ? b0 : 0.0f;
                b1 = k1 + 1 < K ? b1 : 0.0f;
                af.h[q]     = __builtin_amdgcn_cvt_pkrtz(a0, a1);
                af.h[4 + q] = __builtin_amdgcn_cvt_pkrtz(b0, b1);
            }
        }
#pragma unroll
        for (int nt = 0; nt < NT; ++nt) {
            v16h b = *(const v16h*)&Bs[(((kt * NT + nt) * 32 + lane) << 4)];
            acc[nt] = __builtin_amdgcn_wmma_f32_16x16x32_f16(
                false, af.v, false, b, (short)0, acc[nt], false, false);
        }
    }

    // epilogue: C/D layout — m = r + 8*(lane>>4), n = lane&15 (+16*nt)
    int mBase = row0 + ((lane >> 4) << 3);
    int nBase = lane & 15;
#pragma unroll
    for (int nt = 0; nt < NT; ++nt) {
        int n = nt * 16 + nBase;
        float bv = bias ? bias[n] : 0.0f;
#pragma unroll
        for (int r = 0; r < 8; ++r) {
            int m = mBase + r;
            if (m < M) {
                float v = acc[nt][r] + bv;
                if (act == 1) v = v > 0.0f ? v : 0.0f;
                C[(size_t)m * Nout + n] = v;
            }
        }
    }
}

// ---------------------------------------------------------------------------
// Per-edge fused gather + leaky_relu + dot(att) + segment-max (keyed atomic).
// 16 lanes per edge, float4 per lane. src==nullptr => src id == edge id.
// ---------------------------------------------------------------------------
__global__ __launch_bounds__(256) void edge_logits_k(
    const int* __restrict__ src, const int* __restrict__ dst,
    const float* __restrict__ xl, const float* __restrict__ xr,
    const float* __restrict__ eemb, const float* __restrict__ att,
    float* __restrict__ logits, unsigned* __restrict__ segmax, int E)
{
    int g = blockIdx.x * blockDim.x + threadIdx.x;
    int e = g >> 4, sub = g & 15;
    if (e >= E) return;
    int s = src ? src[e] : e;
    int d = dst[e];
    float4 vl = *(const float4*)(xl + (size_t)s * 64 + sub * 4);
    float4 vr = *(const float4*)(xr + (size_t)d * 64 + sub * 4);
    float mx = vl.x + vr.x, my = vl.y + vr.y, mz = vl.z + vr.z, mw = vl.w + vr.w;
    if (eemb) {
        float4 ve = *(const float4*)(eemb + (size_t)e * 64 + sub * 4);
        mx += ve.x; my += ve.y; mz += ve.z; mw += ve.w;
    }
    mx = mx > 0.0f ? mx : 0.01f * mx;
    my = my > 0.0f ? my : 0.01f * my;
    mz = mz > 0.0f ? mz : 0.01f * mz;
    mw = mw > 0.0f ? mw : 0.01f * mw;
    float4 va = *(const float4*)(att + sub * 4);
    float p = mx * va.x + my * va.y + mz * va.z + mw * va.w;
    p += __shfl_xor(p, 8, 16);
    p += __shfl_xor(p, 4, 16);
    p += __shfl_xor(p, 2, 16);
    p += __shfl_xor(p, 1, 16);
    if (sub == 0) {
        logits[e] = p;
        atomicMax(segmax + d, fkey(p));
    }
}

__global__ __launch_bounds__(256) void edge_exp_k(
    const int* __restrict__ dst, const float* __restrict__ logits,
    const unsigned* __restrict__ segmax, float* __restrict__ segsum,
    float* __restrict__ eexp, int E)
{
    int e = blockIdx.x * blockDim.x + threadIdx.x;
    if (e >= E) return;
    int d = dst[e];
    float w = expf(logits[e] - fdec(segmax[d]));
    eexp[e] = w;
    atomicAdd(segsum + d, w);
}

__global__ __launch_bounds__(256) void edge_scatter_k(
    const int* __restrict__ src, const int* __restrict__ dst,
    const float* __restrict__ xl, const float* __restrict__ eexp,
    const float* __restrict__ segsum, float* __restrict__ acc, int E)
{
    int g = blockIdx.x * blockDim.x + threadIdx.x;
    int e = g >> 4, sub = g & 15;
    if (e >= E) return;
    int s = src ? src[e] : e;
    int d = dst[e];
    float a = eexp[e] / (segsum[d] + 1e-16f);
    float4 vl = *(const float4*)(xl + (size_t)s * 64 + sub * 4);
    float* p = acc + (size_t)d * 64 + sub * 4;
    atomicAdd(p + 0, vl.x * a);
    atomicAdd(p + 1, vl.y * a);
    atomicAdd(p + 2, vl.z * a);
    atomicAdd(p + 3, vl.w * a);
}

__global__ __launch_bounds__(256) void bias_elu_k(
    float* __restrict__ acc, const float* __restrict__ bias, int total)
{
    int i = blockIdx.x * blockDim.x + threadIdx.x;
    if (i >= total) return;
    float v = acc[i] + bias[i & 63];
    acc[i] = v > 0.0f ? v : expm1f(v);
}

__global__ __launch_bounds__(256) void gru_combine_k(
    const float* __restrict__ gi, const float* __restrict__ gh,
    float* __restrict__ x, int M)
{
    int idx = blockIdx.x * blockDim.x + threadIdx.x;
    if (idx >= M * 64) return;
    int n = idx >> 6, f = idx & 63;
    const float* gin = gi + (size_t)n * 192;
    const float* ghn = gh + (size_t)n * 192;
    float r = 1.0f / (1.0f + expf(-(gin[f] + ghn[f])));
    float z = 1.0f / (1.0f + expf(-(gin[64 + f] + ghn[64 + f])));
    float nn = tanhf(gin[128 + f] + r * ghn[128 + f]);
    float h = x[idx];
    float o = (1.0f - z) * nn + z * h;
    x[idx] = o > 0.0f ? o : 0.0f;           // relu(gru(...))
}

// ---------------------------------------------------------------------------
// DiffGroupNorm: s = softmax(x@W+b); BN stats as outer-product reductions.
// ---------------------------------------------------------------------------
__global__ __launch_bounds__(256) void dgn_softmax_k(
    const float* __restrict__ x, const float* __restrict__ W,
    const float* __restrict__ b, float* __restrict__ s, int M)
{
    __shared__ float Ws[650];
    for (int i = threadIdx.x; i < 650; i += blockDim.x)
        Ws[i] = (i < 640) ? W[i] : b[i - 640];
    __syncthreads();
    int n = blockIdx.x * blockDim.x + threadIdx.x;
    if (n >= M) return;
    const float* xr = x + (size_t)n * 64;
    float lg[GRP];
#pragma unroll
    for (int g = 0; g < GRP; ++g) lg[g] = Ws[640 + g];
    for (int f = 0; f < 64; ++f) {
        float xv = xr[f];
#pragma unroll
        for (int g = 0; g < GRP; ++g) lg[g] += xv * Ws[f * GRP + g];
    }
    float mx = lg[0];
#pragma unroll
    for (int g = 1; g < GRP; ++g) mx = fmaxf(mx, lg[g]);
    float sum = 0.0f;
#pragma unroll
    for (int g = 0; g < GRP; ++g) { lg[g] = expf(lg[g] - mx); sum += lg[g]; }
    float inv = 1.0f / sum;
#pragma unroll
    for (int g = 0; g < GRP; ++g) s[(size_t)n * GRP + g] = lg[g] * inv;
}

// colsum[g*64+f] = sum_n s[n,g]*x[n,f]; colsq = sum of squares. Wave per node.
__global__ __launch_bounds__(256) void dgn_colstats_k(
    const float* __restrict__ x, const float* __restrict__ s,
    float* __restrict__ colsum, float* __restrict__ colsq, int M)
{
    int lane = threadIdx.x & 31;
    int wave = (blockIdx.x * blockDim.x + threadIdx.x) >> 5;
    int nWaves = (gridDim.x * blockDim.x) >> 5;
    float a0[GRP], a1[GRP], q0[GRP], q1[GRP];
#pragma unroll
    for (int g = 0; g < GRP; ++g) { a0[g] = a1[g] = q0[g] = q1[g] = 0.0f; }
    for (int n = wave; n < M; n += nWaves) {
        float x0 = x[(size_t)n * 64 + lane];
        float x1 = x[(size_t)n * 64 + 32 + lane];
#pragma unroll
        for (int g = 0; g < GRP; ++g) {
            float sg = s[(size_t)n * GRP + g];
            float v0 = sg * x0, v1 = sg * x1;
            a0[g] += v0; a1[g] += v1; q0[g] += v0 * v0; q1[g] += v1 * v1;
        }
    }
#pragma unroll
    for (int g = 0; g < GRP; ++g) {
        atomicAdd(colsum + g * 64 + lane,      a0[g]);
        atomicAdd(colsum + g * 64 + 32 + lane, a1[g]);
        atomicAdd(colsq  + g * 64 + lane,      q0[g]);
        atomicAdd(colsq  + g * 64 + 32 + lane, q1[g]);
    }
}

__global__ __launch_bounds__(256) void dgn_apply_k(
    float* __restrict__ x, const float* __restrict__ s,
    const float* __restrict__ colsum, const float* __restrict__ colsq,
    const float* __restrict__ bw, const float* __restrict__ bb, int M)
{
    int idx = blockIdx.x * blockDim.x + threadIdx.x;
    if (idx >= M * 64) return;
    int n = idx >> 6, f = idx & 63;
    float xv = x[idx];
    float invN = 1.0f / (float)M;
    float accv = 0.0f;
#pragma unroll
    for (int g = 0; g < GRP; ++g) {
        int c = g * 64 + f;
        float mean = colsum[c] * invN;
        float var = colsq[c] * invN - mean * mean;
        float v = s[(size_t)n * GRP + g] * xv;
        accv += (v - mean) * rsqrtf(var + 1e-5f) * bw[c] + bb[c];
    }
    x[idx] = xv + 0.01f * accv;
}

// ---------------------------------------------------------------------------
// Readout helpers
// ---------------------------------------------------------------------------
__global__ __launch_bounds__(256) void seg_feat_sum_k(
    const float* __restrict__ x, const int* __restrict__ batch,
    float* __restrict__ out, int M)
{
    int g = blockIdx.x * blockDim.x + threadIdx.x;
    int n = g >> 4, sub = g & 15;
    if (n >= M) return;
    int b = batch[n];
    float4 v = *(const float4*)(x + (size_t)n * 64 + sub * 4);
    float* p = out + (size_t)b * 64 + sub * 4;
    atomicAdd(p + 0, v.x);
    atomicAdd(p + 1, v.y);
    atomicAdd(p + 2, v.z);
    atomicAdd(p + 3, v.w);
}

__global__ __launch_bounds__(256) void relu_ip_k(float* __restrict__ x, int total)
{
    int i = blockIdx.x * blockDim.x + threadIdx.x;
    if (i >= total) return;
    x[i] = fmaxf(x[i], 0.0f);
}

__global__ __launch_bounds__(256) void head_k(
    const float* __restrict__ y, const float* __restrict__ W,
    const float* __restrict__ b, float* __restrict__ out, int B)
{
    int i = blockIdx.x * blockDim.x + threadIdx.x;
    if (i >= B) return;
    float acc = b[0];
    const float* yr = y + (size_t)i * 64;
#pragma unroll 8
    for (int f = 0; f < 64; ++f) acc += yr[f] * W[f];
    out[i] = acc;
}

// ---------------------------------------------------------------------------
extern "C" void kernel_launch(void* const* d_in, const int* in_sizes, int n_in,
                              void* d_out, int out_size, void* d_ws, size_t ws_size,
                              hipStream_t stream)
{
    (void)in_sizes; (void)n_in; (void)out_size; (void)ws_size;

    const float* x_in    = (const float*)d_in[0];
    const int*   src     = (const int*)  d_in[1];
    const int*   dst     = src + NE;
    const float* eattr   = (const float*)d_in[2];
    const int*   batch   = (const int*)  d_in[3];
    const float* pre_nW  = (const float*)d_in[4];
    const float* pre_nb  = (const float*)d_in[5];
    const float* pre_eW  = (const float*)d_in[6];
    const float* pre_eb  = (const float*)d_in[7];
    const float* g_Wl    = (const float*)d_in[8];
    const float* g_Wl_b  = (const float*)d_in[9];
    const float* g_Wr    = (const float*)d_in[10];
    const float* g_Wr_b  = (const float*)d_in[11];
    const float* g_We    = (const float*)d_in[12];
    const float* g_att   = (const float*)d_in[13];
    const float* g_bias  = (const float*)d_in[14];
    const float* gru_Wih = (const float*)d_in[15];
    const float* gru_Whh = (const float*)d_in[16];
    const float* gru_bih = (const float*)d_in[17];
    const float* gru_bhh = (const float*)d_in[18];
    const float* gn_W    = (const float*)d_in[19];
    const float* gn_b    = (const float*)d_in[20];
    const float* bn_w    = (const float*)d_in[21];
    const float* bn_b    = (const float*)d_in[22];
    const float* ggat_Wl   = (const float*)d_in[23];
    const float* ggat_Wl_b = (const float*)d_in[24];
    const float* ggat_Wr   = (const float*)d_in[25];
    const float* ggat_Wr_b = (const float*)d_in[26];
    const float* ggat_att  = (const float*)d_in[27];
    const float* ggat_bias = (const float*)d_in[28];
    const float* ggru_Wih  = (const float*)d_in[29];
    const float* ggru_Whh  = (const float*)d_in[30];
    const float* ggru_bih  = (const float*)d_in[31];
    const float* ggru_bhh  = (const float*)d_in[32];
    const float* ggn_W     = (const float*)d_in[33];
    const float* ggn_b     = (const float*)d_in[34];
    const float* gbn_w     = (const float*)d_in[35];
    const float* gbn_b     = (const float*)d_in[36];
    const float* post_W    = (const float*)d_in[37];
    const float* post_b    = (const float*)d_in[38];
    const float* out_W     = (const float*)d_in[39];
    const float* out_b     = (const float*)d_in[40];

    // ---- carve workspace ----
    char* base = (char*)d_ws;
    size_t off = 0;
    auto carve = [&](size_t nElems) -> float* {
        float* p = (float*)(base + off);
        off += (nElems * 4 + 255) & ~(size_t)255;
        return p;
    };
    float*    xbuf    = carve((size_t)NN * 64);
    float*    ebuf    = carve((size_t)NE * 64);
    float*    eembbuf = carve((size_t)NE * 64);
    float*    xlbuf   = carve((size_t)NN * 64);
    float*    xrbuf   = carve((size_t)NN * 64);
    float*    accbuf  = carve((size_t)NN * 64);
    float*    gibuf   = carve((size_t)NN * 192);
    float*    ghbuf   = carve((size_t)NN * 192);
    float*    logitb  = carve(NE);
    float*    eexpb   = carve(NE);
    unsigned* segmax  = (unsigned*)carve(NN);
    float*    segsum  = carve(NN);
    float*    sbuf    = carve((size_t)NN * GRP);
    float*    colsum  = carve(640);
    float*    colsq   = carve(640);
    float*    gout    = carve((size_t)NG * 64);
    float*    ybuf    = carve((size_t)NG * 64);
    float*    ybuf2   = carve((size_t)NG * 64);

    auto g64 = [&](const float* A, const float* W, const float* b, float* C,
                   int M, int act) {
        gemm_wmma<4, 64><<<(M + 63) / 64, 128, 0, stream>>>(A, W, b, C, M, act, 0);
    };
    auto g192 = [&](const float* A, const float* W, const float* b, float* C, int M) {
        gemm_wmma<12, 64><<<(M + 63) / 64, 128, 0, stream>>>(A, W, b, C, M, 0, 1);
    };

    // ---- pre-embeddings (odd K: 92 and 50, tail tile handled branch-free) ----
    gemm_wmma<4, 92><<<(NN + 63) / 64, 128, 0, stream>>>(x_in,  pre_nW, pre_nb, xbuf, NN, 1, 0);
    gemm_wmma<4, 50><<<(NE + 63) / 64, 128, 0, stream>>>(eattr, pre_eW, pre_eb, ebuf, NE, 1, 0);

    // ---- attention blocks ----
    for (int l = 0; l < 2; ++l) {
        g64(ebuf, g_We + l * 4096, nullptr, eembbuf, NE, 0);
        for (int t = 0; t < 2; ++t) {
            g64(xbuf, g_Wl + l * 4096, g_Wl_b + l * 64, xlbuf, NN, 0);
            g64(xbuf, g_Wr + l * 4096, g_Wr_b + l * 64, xrbuf, NN, 0);
            (void)hipMemsetAsync(segmax, 0, (size_t)NN * 4, stream);
            (void)hipMemsetAsync(segsum, 0, (size_t)NN * 4, stream);
            (void)hipMemsetAsync(accbuf, 0, (size_t)NN * 64 * 4, stream);
            int eb = (NE * 16 + 255) / 256;
            edge_logits_k<<<eb, 256, 0, stream>>>(src, dst, xlbuf, xrbuf,
                eembbuf, g_att + l * 64, logitb, segmax, NE);
            edge_exp_k<<<(NE + 255) / 256, 256, 0, stream>>>(dst, logitb,
                segmax, segsum, eexpb, NE);
            edge_scatter_k<<<eb, 256, 0, stream>>>(src, dst, xlbuf, eexpb,
                segsum, accbuf, NE);
            bias_elu_k<<<(NN * 64 + 255) / 256, 256, 0, stream>>>(accbuf,
                g_bias + l * 64, NN * 64);
            g192(accbuf, gru_Wih + l * 192 * 64, gru_bih + l * 192, gibuf, NN);
            g192(xbuf,   gru_Whh + l * 192 * 64, gru_bhh + l * 192, ghbuf, NN);
            gru_combine_k<<<(NN * 64 + 255) / 256, 256, 0, stream>>>(gibuf, ghbuf, xbuf, NN);
        }
        dgn_softmax_k<<<(NN + 255) / 256, 256, 0, stream>>>(xbuf,
            gn_W + l * 640, gn_b + l * GRP, sbuf, NN);
        (void)hipMemsetAsync(colsum, 0, 640 * 4, stream);
        (void)hipMemsetAsync(colsq,  0, 640 * 4, stream);
        dgn_colstats_k<<<512, 256, 0, stream>>>(xbuf, sbuf, colsum, colsq, NN);
        dgn_apply_k<<<(NN * 64 + 255) / 256, 256, 0, stream>>>(xbuf, sbuf,
            colsum, colsq, bn_w + l * 640, bn_b + l * 640, NN);
    }

    // ---- global readout ----
    (void)hipMemsetAsync(gout, 0, (size_t)NG * 64 * 4, stream);
    seg_feat_sum_k<<<(NN * 16 + 255) / 256, 256, 0, stream>>>(xbuf, batch, gout, NN);
    relu_ip_k<<<(NG * 64 + 255) / 256, 256, 0, stream>>>(gout, NG * 64);

    // xl is loop-invariant in the readout loop (x fixed)
    g64(xbuf, ggat_Wl, ggat_Wl_b, xlbuf, NN, 0);
    int nb16 = (NN * 16 + 255) / 256;
    for (int t = 0; t < 2; ++t) {
        g64(gout, ggat_Wr, ggat_Wr_b, xrbuf, NG, 0);
        (void)hipMemsetAsync(segmax, 0, (size_t)NG * 4, stream);
        (void)hipMemsetAsync(segsum, 0, (size_t)NG * 4, stream);
        (void)hipMemsetAsync(accbuf, 0, (size_t)NG * 64 * 4, stream);
        edge_logits_k<<<nb16, 256, 0, stream>>>((const int*)nullptr, batch,
            xlbuf, xrbuf, (const float*)nullptr, ggat_att, logitb, segmax, NN);
        edge_exp_k<<<(NN + 255) / 256, 256, 0, stream>>>(batch, logitb,
            segmax, segsum, eexpb, NN);
        edge_scatter_k<<<nb16, 256, 0, stream>>>((const int*)nullptr, batch,
            xlbuf, eexpb, segsum, accbuf, NN);
        bias_elu_k<<<(NG * 64 + 255) / 256, 256, 0, stream>>>(accbuf, ggat_bias, NG * 64);
        g192(accbuf, ggru_Wih, ggru_bih, gibuf, NG);
        g192(gout,   ggru_Whh, ggru_bhh, ghbuf, NG);
        gru_combine_k<<<(NG * 64 + 255) / 256, 256, 0, stream>>>(gibuf, ghbuf, gout, NG);
    }

    // ---- graph-level DiffGroupNorm ----
    dgn_softmax_k<<<(NG + 255) / 256, 256, 0, stream>>>(gout, ggn_W, ggn_b, sbuf, NG);
    (void)hipMemsetAsync(colsum, 0, 640 * 4, stream);
    (void)hipMemsetAsync(colsq,  0, 640 * 4, stream);
    dgn_colstats_k<<<64, 256, 0, stream>>>(gout, sbuf, colsum, colsq, NG);
    dgn_apply_k<<<(NG * 64 + 255) / 256, 256, 0, stream>>>(gout, sbuf,
        colsum, colsq, gbn_w, gbn_b, NG);

    // ---- post MLP + head ----
    g64(gout, post_W,        post_b,      ybuf,  NG, 1);
    g64(ybuf, post_W + 4096, post_b + 64, ybuf2, NG, 1);
    head_k<<<(NG + 255) / 256, 256, 0, stream>>>(ybuf2, out_W, out_b, (float*)d_out, NG);
}
